// DSMoE_47004122087941
// MI455X (gfx1250) — compile-verified, matrix operand order checked by reading, not run on hardware
//
#include <hip/hip_runtime.h>
#include <hip/hip_bf16.h>

// ---------------------------------------------------------------------------
// MoE FFN for MI455X (gfx1250): bf16 WMMA, sparse top-2 dispatch.
//   N = 16384 tokens, C = 512, FF = 2048, E = 4, TOP_K = 2
// d_out = [ out (N*512 f32) | router_sparse (N*4 f32) ]
// ---------------------------------------------------------------------------

typedef __attribute__((ext_vector_type(16))) __bf16 v16bf;
typedef __attribute__((ext_vector_type(8)))  float  v8f;

union FragBF {
    v16bf v;
    uint4 q[2];
};

#define NTOK   16384
#define NC     512
#define NFF    2048
#define NEXP   4

// ---------------- workspace layout (bytes) ----------------
// [0,256)            : int counters[4]
// [256, +256KB)      : int   toklist[4][16384]
// [.. , +256KB)      : float wtlist[4][16384]
// [1MB, 17MB)        : bf16  xbf[16384][512]
// [17MB, 25MB)       : bf16  w1t[4][2048][512]   (= w1 transposed, [ff][c])
// [25MB, 33MB)       : bf16  w2t[4][512][2048]   (= w2 transposed, [c][ff])

// ---------------------------------------------------------------------------
// Weight convert + transpose kernels (strided reads absorbed by 192MB L2).
// ---------------------------------------------------------------------------
__global__ void conv_w1(const float* __restrict__ w1, __bf16* __restrict__ w1t) {
    int o = blockIdx.x * blockDim.x + threadIdx.x;      // 4 * 2048 * 512 = 1<<22
    if (o >= NEXP * NFF * NC) return;
    int e   = o >> 20;                                  // / (2048*512)
    int rem = o & ((1 << 20) - 1);
    int ff  = rem >> 9;                                 // / 512
    int c   = rem & 511;
    w1t[o] = (__bf16)w1[(e * NC + c) * NFF + ff];
}

__global__ void conv_w2(const float* __restrict__ w2, __bf16* __restrict__ w2t) {
    int o = blockIdx.x * blockDim.x + threadIdx.x;
    if (o >= NEXP * NC * NFF) return;
    int e   = o >> 20;
    int rem = o & ((1 << 20) - 1);
    int c   = rem >> 11;                                // / 2048
    int ff  = rem & 2047;
    w2t[o] = (__bf16)w2[(e * NFF + ff) * NC + c];
}

// ---------------------------------------------------------------------------
// Routing: one wave per token. Gate logits + softmax + top-2 + compaction,
// plus f32 -> bf16 conversion of the token row.
// ---------------------------------------------------------------------------
__global__ void moe_route(const float* __restrict__ x,
                          const float* __restrict__ gate_w,
                          float* __restrict__ router_out,
                          __bf16* __restrict__ xbf,
                          int* __restrict__ cnts,
                          int* __restrict__ toklist,
                          float* __restrict__ wtlist) {
    const int wv = threadIdx.x >> 5;
    const int ln = threadIdx.x & 31;
    const int n  = blockIdx.x * 8 + wv;
    const int c0 = ln * 16;

    const float4* xr = (const float4*)(x + (size_t)n * NC + c0);
    const float4* gw = (const float4*)gate_w;           // gate_w[c] is a float4 of 4 experts

    float4 acc = {0.f, 0.f, 0.f, 0.f};
    union { __bf16 b[16]; uint4 q[2]; } u;
#pragma unroll
    for (int i = 0; i < 4; ++i) {
        float4 xv = xr[i];
        float xs[4] = {xv.x, xv.y, xv.z, xv.w};
#pragma unroll
        for (int k = 0; k < 4; ++k) {
            float  xc = xs[k];
            float4 g  = gw[c0 + i * 4 + k];
            acc.x += xc * g.x; acc.y += xc * g.y;
            acc.z += xc * g.z; acc.w += xc * g.w;
            u.b[i * 4 + k] = (__bf16)xc;
        }
    }
    // store bf16 token row
    uint4* xo = (uint4*)(xbf + (size_t)n * NC + c0);
    xo[0] = u.q[0]; xo[1] = u.q[1];

    // wave32 xor-reduction of the 4 logits
#pragma unroll
    for (int off = 16; off > 0; off >>= 1) {
        acc.x += __shfl_xor(acc.x, off, 32);
        acc.y += __shfl_xor(acc.y, off, 32);
        acc.z += __shfl_xor(acc.z, off, 32);
        acc.w += __shfl_xor(acc.w, off, 32);
    }

    float l[4] = {acc.x, acc.y, acc.z, acc.w};
    float m = fmaxf(fmaxf(l[0], l[1]), fmaxf(l[2], l[3]));
    float p[4];
#pragma unroll
    for (int e = 0; e < 4; ++e) p[e] = expf(l[e] - m);
    // top-2, lowest index wins on ties (matches jax top_k)
    int i0 = 0;
#pragma unroll
    for (int e = 1; e < 4; ++e) if (p[e] > p[i0]) i0 = e;
    int i1 = (i0 == 0) ? 1 : 0;
#pragma unroll
    for (int e = 0; e < 4; ++e) if (e != i0 && p[e] > p[i1]) i1 = e;
    float s   = p[i0] + p[i1];
    float w0  = p[i0] / s;
    float w1v = p[i1] / s;

    if (ln == 0) {
        float r4[4] = {0.f, 0.f, 0.f, 0.f};
        r4[i0] = w0; r4[i1] = w1v;
        float4 rv = {r4[0], r4[1], r4[2], r4[3]};
        ((float4*)router_out)[n] = rv;
        int p0 = atomicAdd(&cnts[i0], 1);
        toklist[i0 * NTOK + p0] = n;
        wtlist[i0 * NTOK + p0] = w0;
        int p1 = atomicAdd(&cnts[i1], 1);
        toklist[i1 * NTOK + p1] = n;
        wtlist[i1 * NTOK + p1] = w1v;
    }
}

// ---------------------------------------------------------------------------
// Expert MLP: per (expert, 64-token tile). 256 threads = 8 waves.
// Phase A: h[64,64] = gelu(x_tile @ w1[:, ff0:ff0+64])  (bf16 WMMA, f32 acc)
// Phase B: y[64,512] += h @ w2[ff0:ff0+64, :]           (bf16 WMMA, f32 acc)
// Epilogue: atomic scatter-add of weight * y into out.
// ---------------------------------------------------------------------------
__global__ void __launch_bounds__(256)
moe_expert_mlp(const __bf16* __restrict__ xbf,
               const __bf16* __restrict__ w1t,
               const __bf16* __restrict__ w2t,
               const int* __restrict__ cnts,
               const int* __restrict__ toklist,
               const float* __restrict__ wtlist,
               float* __restrict__ out) {
    const int e        = blockIdx.y;
    const int cnt      = cnts[e];
    const int tilebase = blockIdx.x * 64;
    if (tilebase >= cnt) return;                        // uniform early-exit

    __shared__ int    sh_tok[64];
    __shared__ float  sh_w[64];
    __shared__ __align__(16) __bf16 hbuf[64 * 64];      // 8 KB, b128-aligned

    const int t = threadIdx.x;
    if (t < 64) {
        int idx = tilebase + t;
        if (idx < cnt) {
            sh_tok[t] = toklist[e * NTOK + idx];
            sh_w[t]   = wtlist[e * NTOK + idx];
        } else {
            sh_tok[t] = 0;                              // padded slot: token 0, weight 0
            sh_w[t]   = 0.0f;
        }
    }
    __syncthreads();

    const int wv  = t >> 5;
    const int ln  = t & 31;
    const int lh  = ln >> 4;                            // lane half
    const int l15 = ln & 15;
    const int kb8 = lh * 8;                             // K offset within half

    // phase-A work split: wave -> (ff 16-col group, 2 m-tiles)
    const int fgrp  = wv & 3;
    const int mbase = (wv >> 2) * 2;
    const __bf16* xrow[2];
#pragma unroll
    for (int j = 0; j < 2; ++j) {
        int row = (mbase + j) * 16 + l15;
        xrow[j] = xbf + (size_t)sh_tok[row] * NC;
    }
    const __bf16* w1e = w1t + (size_t)e * NFF * NC;
    const __bf16* w2e = w2t + (size_t)e * NC * NFF;

    const v8f vzero = {0.f, 0.f, 0.f, 0.f, 0.f, 0.f, 0.f, 0.f};
    v8f acc[4][4];
#pragma unroll
    for (int mt = 0; mt < 4; ++mt)
#pragma unroll
        for (int ct = 0; ct < 4; ++ct) acc[mt][ct] = vzero;

    for (int ff0 = 0; ff0 < NFF; ff0 += 64) {
        // ---------------- Phase A: h tile ----------------
        v8f ah[2] = {vzero, vzero};
        const int ffcol = ff0 + fgrp * 16 + l15;
        const __bf16* brow = w1e + (size_t)ffcol * NC;
#pragma unroll
        for (int ks = 0; ks < NC; ks += 32) {
            FragBF bf;
            const uint4* bp = (const uint4*)(brow + ks + kb8);
            bf.q[0] = bp[0]; bf.q[1] = bp[2];
#pragma unroll
            for (int j = 0; j < 2; ++j) {
                FragBF af;
                const uint4* ap = (const uint4*)(xrow[j] + ks + kb8);
                af.q[0] = ap[0]; af.q[1] = ap[2];
                ah[j] = __builtin_amdgcn_wmma_f32_16x16x32_bf16(
                    false, af.v, false, bf.v, (short)0, ah[j], false, false);
            }
        }
        // exact GELU (erf form) + pack bf16 into LDS hbuf[m][ff]
#pragma unroll
        for (int j = 0; j < 2; ++j) {
            int mt = mbase + j;
#pragma unroll
            for (int r = 0; r < 8; ++r) {
                float v = ah[j][r];
                float g = 0.5f * v * (1.0f + erff(v * 0.70710678118654752f));
                int mrow = mt * 16 + r + lh * 8;
                hbuf[mrow * 64 + fgrp * 16 + l15] = (__bf16)g;
            }
        }
        __syncthreads();

        // ---------------- Phase B: y += h @ w2 chunk ----------------
        const int c0 = wv * 64;
#pragma unroll
        for (int kc = 0; kc < 2; ++kc) {
            FragBF bfr[4];
#pragma unroll
            for (int ct = 0; ct < 4; ++ct) {
                int col = c0 + ct * 16 + l15;
                const uint4* bp =
                    (const uint4*)(w2e + (size_t)col * NFF + ff0 + kc * 32 + kb8);
                bfr[ct].q[0] = bp[0]; bfr[ct].q[1] = bp[2];
            }
#pragma unroll
            for (int mt = 0; mt < 4; ++mt) {
                FragBF af;
                const uint4* ap =
                    (const uint4*)(&hbuf[(mt * 16 + l15) * 64 + kc * 32 + kb8]);
                af.q[0] = ap[0]; af.q[1] = ap[2];
#pragma unroll
                for (int ct = 0; ct < 4; ++ct)
                    acc[mt][ct] = __builtin_amdgcn_wmma_f32_16x16x32_bf16(
                        false, af.v, false, bfr[ct].v, (short)0, acc[mt][ct],
                        false, false);
            }
        }
        __syncthreads();
    }

    // ---------------- Epilogue: weighted scatter-add ----------------
#pragma unroll
    for (int mt = 0; mt < 4; ++mt)
#pragma unroll
        for (int ct = 0; ct < 4; ++ct)
#pragma unroll
            for (int r = 0; r < 8; ++r) {
                int mrow = mt * 16 + r + lh * 8;
                int col  = wv * 64 + ct * 16 + l15;
                float v  = acc[mt][ct][r] * sh_w[mrow];
                unsafeAtomicAdd(out + (size_t)sh_tok[mrow] * NC + col, v);
            }
}

// ---------------------------------------------------------------------------
extern "C" void kernel_launch(void* const* d_in, const int* in_sizes, int n_in,
                              void* d_out, int out_size, void* d_ws, size_t ws_size,
                              hipStream_t stream) {
    const float* x      = (const float*)d_in[0];
    const float* gate_w = (const float*)d_in[1];
    const float* w1     = (const float*)d_in[2];
    const float* w2     = (const float*)d_in[3];

    float* out    = (float*)d_out;
    float* router = out + (size_t)NTOK * NC;

    char* ws = (char*)d_ws;
    int*    cnts    = (int*)ws;
    int*    toklist = (int*)(ws + 256);
    float*  wtlist  = (float*)(ws + 256 + (size_t)NEXP * NTOK * 4);
    __bf16* xbf     = (__bf16*)(ws + (1u << 20));
    __bf16* w1t     = (__bf16*)(ws + (1u << 20) + (16u << 20));
    __bf16* w2t     = (__bf16*)(ws + (1u << 20) + (24u << 20));

    hipMemsetAsync(cnts, 0, 256, stream);
    hipMemsetAsync(out, 0, (size_t)NTOK * NC * sizeof(float), stream);

    conv_w1<<<16384, 256, 0, stream>>>(w1, w1t);
    conv_w2<<<16384, 256, 0, stream>>>(w2, w2t);
    moe_route<<<NTOK / 8, 256, 0, stream>>>(x, gate_w, router, xbf, cnts,
                                            toklist, wtlist);
    moe_expert_mlp<<<dim3(256, NEXP), 256, 0, stream>>>(xbf, w1t, w2t, cnts,
                                                        toklist, wtlist, out);
}
